// Seq2Seq_24876450578597
// MI455X (gfx1250) — matrix operand order repaired and checked by reading, not compiled, not run on hardware
//
#include <hip/hip_runtime.h>
#include <math.h>

typedef unsigned short ushort_t;
typedef unsigned int   uint32;
typedef __attribute__((ext_vector_type(16))) __bf16 v16bf;
typedef __attribute__((ext_vector_type(8)))  __bf16 v8bf;
typedef __attribute__((ext_vector_type(8)))  float  v8f;

#define HDIM   512
#define EDIM   512
#define G4     2048          // 4*H
#define VTRG   32000
#define SEQ    512
#define TDEC   511
#define NCHUNK 25
#define CHUNKW 1280
#define SCAN_WG 64

// ---------------- helpers ----------------

__device__ __forceinline__ ushort_t f2bf(float f) {
  uint32 u = __float_as_uint(f);
  u += 0x7FFFu + ((u >> 16) & 1u);     // round to nearest even
  return (ushort_t)(u >> 16);
}
__device__ __forceinline__ float bf2f(ushort_t u) {
  return __uint_as_float(((uint32)u) << 16);
}

// Per-lane async copy of one 16B chunk from global to LDS (ASYNCcnt-tracked).
// LDS byte offset = low 32 bits of the generic shared pointer.
__device__ __forceinline__ void async_g2l_b128(const void* gptr, void* lptr) {
  uint32 lds = (uint32)(unsigned long long)lptr;
  unsigned long long ga = (unsigned long long)gptr;
  asm volatile("global_load_async_to_lds_b128 %0, %1, off"
               :: "v"(lds), "v"(ga) : "memory");
}
__device__ __forceinline__ void async_wait_all() {
  asm volatile("s_wait_asynccnt 0" ::: "memory");
}

// A fragment for v_wmma_f32_16x16x32_bf16 (works on global or LDS pointers).
// Layout (16-bit A 16x32, contiguous K to match B): lanes 0-15 hold M=lane, K=0..15;
// lanes 16-31 hold M=lane-16, K=16..31.
__device__ __forceinline__ v16bf load_a_frag(const ushort_t* arow, int k, int sel8) {
  return *(const v16bf*)(arow + k + 2 * sel8);
}

// Stage a 16-row x 512-col bf16 A tile into LDS via async loads.
// 16*512*2B = 16KB = 1024 chunks of 16B; 256 threads x 4 iterations.
__device__ __forceinline__ void stage_a_tile(const ushort_t* __restrict__ A,
                                             int m0, ushort_t* smem, int tid) {
  for (int c = tid; c < 16 * (HDIM / 8); c += 256) {
    int row = c >> 6;            // 64 chunks per 512-elem row
    int k8  = (c & 63) * 8;
    async_g2l_b128(A + (size_t)(m0 + row) * HDIM + k8, smem + row * HDIM + k8);
  }
  async_wait_all();
  __syncthreads();
}

// Sense-reversing grid barrier (all participating workgroups are co-resident).
__device__ __forceinline__ void grid_barrier(int* bar, int nwg, int tid) {
  __syncthreads();
  if (tid == 0) {
    volatile int* vgen = bar + 1;
    int g = *vgen;
    __threadfence();
    if (atomicAdd(bar, 1) == nwg - 1) {
      *bar = 0;
      __threadfence();
      atomicAdd(bar + 1, 1);
    } else {
      while (*vgen == g) { __builtin_amdgcn_s_sleep(1); }
    }
    __threadfence();
  }
  __syncthreads();
}

// ---------------- small utility kernels ----------------

__global__ void k_zero_i32(int* p, int n) {
  int i = blockIdx.x * blockDim.x + threadIdx.x;
  if (i < n) p[i] = 0;
}

__global__ void k_f32_to_bf16(const float* __restrict__ src,
                              ushort_t* __restrict__ dst, int n) {
  int i = blockIdx.x * blockDim.x + threadIdx.x;
  int stride = gridDim.x * blockDim.x;
  for (; i < n; i += stride) dst[i] = f2bf(src[i]);
}

// Gather embedding rows by token id, convert to bf16, zero-pad rows >= T.
__global__ void k_gather_embed(const float* __restrict__ embed,
                               const int* __restrict__ idx,
                               ushort_t* __restrict__ out, int T) {
  int row = blockIdx.x;
  ushort_t* o = out + (size_t)row * EDIM;
  if (row < T) {
    const float* src = embed + (size_t)idx[row] * EDIM;
    for (int e = threadIdx.x; e < EDIM; e += blockDim.x) o[e] = f2bf(src[e]);
  } else {
    for (int e = threadIdx.x; e < EDIM; e += blockDim.x) o[e] = 0;
  }
}

// hs (M x 512 fp32) -> bf16, zero-pad rows >= M up to 512.
__global__ void k_hs_to_bf16(const float* __restrict__ hs,
                             ushort_t* __restrict__ out, int M) {
  int row = blockIdx.x;
  ushort_t* o = out + (size_t)row * HDIM;
  if (row < M) {
    const float* src = hs + (size_t)row * HDIM;
    for (int e = threadIdx.x; e < HDIM; e += blockDim.x) o[e] = f2bf(src[e]);
  } else {
    for (int e = threadIdx.x; e < HDIM; e += blockDim.x) o[e] = 0;
  }
}

// ---------------- WMMA GEMM: C[m][n] = sum_k A[m][k]*B[n][k] + b1[n] + b2[n] ----------------
// A: Mp x 512 bf16 row-major (Mp multiple of 16), B: N x 512 bf16 row-major.
// grid = (N/128, Mp/16), block = 256 (8 waves, each owns a 16x16 tile).
// A tile staged to LDS once per workgroup via async-to-LDS; B streamed global.

__global__ void k_wmma_gemm_bias(const ushort_t* __restrict__ A,
                                 const ushort_t* __restrict__ B,
                                 const float* __restrict__ b1,
                                 const float* __restrict__ b2,
                                 float* __restrict__ C, int N) {
  __shared__ __align__(16) ushort_t a_s[16 * HDIM];   // 16 KB
  int lane = threadIdx.x & 31;
  int wave = threadIdx.x >> 5;
  int m0 = blockIdx.y * 16;
  int n0 = blockIdx.x * 128 + wave * 16;
  int sel  = lane >> 4;
  int sel8 = sel * 8;

  stage_a_tile(A, m0, a_s, threadIdx.x);

  const ushort_t* arow = a_s + (size_t)(lane & 15) * HDIM;
  const ushort_t* brow = B + (size_t)(n0 + (lane & 15)) * HDIM + sel * 16;
  v8f acc = {};
  for (int k = 0; k < HDIM; k += 32) {
    v16bf a = load_a_frag(arow, k, sel8);
    v16bf b = *(const v16bf*)(brow + k);
    acc = __builtin_amdgcn_wmma_f32_16x16x32_bf16(false, a, false, b,
                                                  (short)0, acc, false, false);
  }
  int n = n0 + (lane & 15);
  float bias = (b1 ? b1[n] : 0.f) + (b2 ? b2[n] : 0.f);
  int mbase = m0 + sel * 8;
#pragma unroll
  for (int r = 0; r < 8; ++r)
    C[(size_t)(mbase + r) * N + n] = acc[r] + bias;
}

// ---------------- Persistent LSTM scan ----------------
// 64 workgroups x 256 threads. WG g owns h[j], j in [g*8, g*8+8), and keeps the
// 32 corresponding Whh rows (i,f,g,o gates) resident in LDS as bf16 (32 KB,
// staged via async-to-LDS). h is exchanged via a double-buffered global vector
// + grid barrier per step.

__global__ void k_lstm_scan(const float* __restrict__ pre,
                            const ushort_t* __restrict__ Whh,
                            const float* __restrict__ h_in,
                            const float* __restrict__ c_in,
                            int T,
                            float* __restrict__ hs_out,
                            float* __restrict__ h_out,
                            float* __restrict__ c_out,
                            float* __restrict__ hbuf,
                            int* __restrict__ bar) {
  __shared__ __align__(16) ushort_t whh_s[32 * HDIM];   // 32 KB
  __shared__ float h_s[HDIM];
  __shared__ float part[32][9];
  __shared__ float gate_s[32];
  __shared__ float c_s[8];

  int tid = threadIdx.x;
  int j0  = blockIdx.x * 8;

  // stage our Whh slice into LDS (bf16) with async copies:
  // 32 rows x 512 = 2048 chunks of 16B, 8 per thread.
  for (int c = tid; c < 32 * (HDIM / 8); c += 256) {
    int lr = c >> 6;                         // 64 chunks per row
    int k8 = (c & 63) * 8;
    int grow = (lr >> 3) * HDIM + j0 + (lr & 7);   // gate row in [0,2048)
    async_g2l_b128(Whh + (size_t)grow * HDIM + k8, whh_s + lr * HDIM + k8);
  }
  if (tid < 8) {
    c_s[tid] = c_in ? c_in[j0 + tid] : 0.f;
    hbuf[j0 + tid] = h_in ? h_in[j0 + tid] : 0.f;  // h_prev buffer 0
  }
  async_wait_all();
  __threadfence();
  grid_barrier(bar, SCAN_WG, tid);

  int lr  = tid >> 3;          // local gate row 0..31
  int sub = tid & 7;           // K-slice 0..7 (64 elements each)
  const ushort_t* wrow = whh_s + lr * HDIM + sub * 64;

  for (int t = 0; t < T; ++t) {
    const float* cur = hbuf + (t & 1) * HDIM;
    float*       nxt = hbuf + ((t + 1) & 1) * HDIM;

    h_s[tid]       = cur[tid];
    h_s[tid + 256] = cur[tid + 256];
    __syncthreads();

    const float* hh = h_s + sub * 64;
    float p = 0.f;
#pragma unroll 8
    for (int k = 0; k < 64; ++k) p += bf2f(wrow[k]) * hh[k];
    part[lr][sub] = p;
    __syncthreads();

    if (tid < 32) {
      float s = 0.f;
#pragma unroll
      for (int q = 0; q < 8; ++q) s += part[tid][q];
      int grow = (tid >> 3) * HDIM + j0 + (tid & 7);
      gate_s[tid] = s + pre[(size_t)t * G4 + grow];
    }
    __syncthreads();

    if (tid < 8) {
      float gi = gate_s[tid];
      float gf = gate_s[8 + tid];
      float gg = gate_s[16 + tid];
      float go = gate_s[24 + tid];
      float i = 1.f / (1.f + __expf(-gi));
      float f = 1.f / (1.f + __expf(-gf));
      float g = tanhf(gg);
      float o = 1.f / (1.f + __expf(-go));
      float c = f * c_s[tid] + i * g;
      c_s[tid] = c;
      float h = o * tanhf(c);
      nxt[j0 + tid] = h;
      if (hs_out) hs_out[(size_t)t * HDIM + j0 + tid] = h;
      if (t == T - 1) { h_out[j0 + tid] = h; c_out[j0 + tid] = c; }
    }
    __threadfence();
    grid_barrier(bar, SCAN_WG, tid);
  }
}

// ---------------- Fused logits GEMM + online logsumexp ----------------
// grid = (25 N-chunks, 32 M-tiles), block = 256 (8 waves).
// A tile (16 rows of hs) staged in LDS once, swept against the 1280-wide
// W_out chunk (10 iterations of 8x16 columns), maintaining per-row running
// (max, sum-exp) in lane registers and capturing the target logit.

__global__ void k_logits_lse(const ushort_t* __restrict__ Abf,   // 512x512 bf16 hs (padded)
                             const ushort_t* __restrict__ Wout,  // 32000x512 bf16
                             const float* __restrict__ b_out,
                             const int* __restrict__ target,
                             float* __restrict__ cmax,
                             float* __restrict__ csum,
                             float* __restrict__ tgt_logit,
                             int M) {
  __shared__ __align__(16) ushort_t a_s[16 * HDIM];   // 16 KB
  __shared__ float tile[8][16][17];
  __shared__ float wmaxs[8][16];
  __shared__ float wsums[8][16];

  int lane  = threadIdx.x & 31;
  int wave  = threadIdx.x >> 5;
  int mtile = blockIdx.y;
  int chunk = blockIdx.x;
  int m0 = mtile * 16;
  int sel = lane >> 4, sel8 = sel * 8;

  stage_a_tile(Abf, m0, a_s, threadIdx.x);
  const ushort_t* arow = a_s + (size_t)(lane & 15) * HDIM;

  float rmax = -INFINITY, rsum = 0.f;
  int myrow = m0 + lane;                 // stats role for lanes 0..15
  int mytgt = -1;
  if (lane < 16 && myrow < M) mytgt = target[myrow + 1];   // trg_out[m]

  for (int it = 0; it < 10; ++it) {
    int n0w = chunk * CHUNKW + it * 128 + wave * 16;
    const ushort_t* brow = Wout + (size_t)(n0w + (lane & 15)) * HDIM + sel * 16;
    v8f acc = {};
    for (int k = 0; k < HDIM; k += 32) {
      v16bf a = load_a_frag(arow, k, sel8);
      v16bf b = *(const v16bf*)(brow + k);
      acc = __builtin_amdgcn_wmma_f32_16x16x32_bf16(false, a, false, b,
                                                    (short)0, acc, false, false);
    }
    float bias = b_out[n0w + (lane & 15)];
    int rbase = sel * 8;
#pragma unroll
    for (int r = 0; r < 8; ++r)
      tile[wave][rbase + r][lane & 15] = acc[r] + bias;
    __syncthreads();

    if (lane < 16) {
      float tmax = -INFINITY;
#pragma unroll
      for (int j = 0; j < 16; ++j) tmax = fmaxf(tmax, tile[wave][lane][j]);
      float nm = fmaxf(rmax, tmax);
      float s = rsum * __expf(rmax - nm);
#pragma unroll
      for (int j = 0; j < 16; ++j) s += __expf(tile[wave][lane][j] - nm);
      rmax = nm; rsum = s;
      if (mytgt >= n0w && mytgt < n0w + 16)
        tgt_logit[myrow] = tile[wave][lane][mytgt - n0w];
    }
    __syncthreads();
  }

  if (lane < 16) { wmaxs[wave][lane] = rmax; wsums[wave][lane] = rsum; }
  __syncthreads();
  if (threadIdx.x < 16) {
    float gm = -INFINITY;
    for (int w = 0; w < 8; ++w) gm = fmaxf(gm, wmaxs[w][threadIdx.x]);
    float gs = 0.f;
    for (int w = 0; w < 8; ++w)
      gs += wsums[w][threadIdx.x] * __expf(wmaxs[w][threadIdx.x] - gm);
    size_t o = ((size_t)mtile * NCHUNK + chunk) * 16 + threadIdx.x;
    cmax[o] = gm; csum[o] = gs;
  }
}

// Merge per-chunk stats into losses: loss[m] = logsumexp(row m) - logit[target].
__global__ void k_loss_final(const float* __restrict__ cmax,
                             const float* __restrict__ csum,
                             const float* __restrict__ tgt_logit,
                             float* __restrict__ out, int M) {
  int m = blockIdx.x * blockDim.x + threadIdx.x;
  if (m >= M) return;
  int mtile = m >> 4, row = m & 15;
  float gm = -INFINITY;
  for (int c = 0; c < NCHUNK; ++c)
    gm = fmaxf(gm, cmax[((size_t)mtile * NCHUNK + c) * 16 + row]);
  float gs = 0.f;
  for (int c = 0; c < NCHUNK; ++c) {
    size_t o = ((size_t)mtile * NCHUNK + c) * 16 + row;
    gs += csum[o] * __expf(cmax[o] - gm);
  }
  out[m] = (gm + logf(gs)) - tgt_logit[m];
}

// ---------------- host orchestration ----------------

extern "C" void kernel_launch(void* const* d_in, const int* in_sizes, int n_in,
                              void* d_out, int out_size, void* d_ws, size_t ws_size,
                              hipStream_t stream) {
  const int*   source  = (const int*)  d_in[0];
  const int*   target  = (const int*)  d_in[1];
  const float* enc_emb = (const float*)d_in[2];
  const float* enc_Wih = (const float*)d_in[3];
  const float* enc_Whh = (const float*)d_in[4];
  const float* enc_bih = (const float*)d_in[5];
  const float* enc_bhh = (const float*)d_in[6];
  const float* dec_emb = (const float*)d_in[7];
  const float* dec_Wih = (const float*)d_in[8];
  const float* dec_Whh = (const float*)d_in[9];
  const float* dec_bih = (const float*)d_in[10];
  const float* dec_bhh = (const float*)d_in[11];
  const float* W_out   = (const float*)d_in[12];
  const float* b_out   = (const float*)d_in[13];
  float* losses = (float*)d_out;

  char* base = (char*)d_ws;
  size_t off = 0;
  auto carve = [&](size_t bytes) -> char* {
    char* p = base + off;
    off = (off + bytes + 255) & ~(size_t)255;
    return p;
  };

  ushort_t* wihE = (ushort_t*)carve((size_t)G4 * EDIM * 2);
  ushort_t* whhE = (ushort_t*)carve((size_t)G4 * HDIM * 2);
  ushort_t* wihD = (ushort_t*)carve((size_t)G4 * EDIM * 2);
  ushort_t* whhD = (ushort_t*)carve((size_t)G4 * HDIM * 2);
  ushort_t* woutB = (ushort_t*)carve((size_t)VTRG * HDIM * 2);
  ushort_t* embE = (ushort_t*)carve((size_t)SEQ * EDIM * 2);
  ushort_t* embD = (ushort_t*)carve((size_t)SEQ * EDIM * 2);
  float* preE = (float*)carve((size_t)SEQ * G4 * 4);
  float* preD = (float*)carve((size_t)SEQ * G4 * 4);
  float* hbuf = (float*)carve(2 * HDIM * 4);
  float* hEnc = (float*)carve(HDIM * 4);
  float* cEnc = (float*)carve(HDIM * 4);
  float* hDec = (float*)carve(HDIM * 4);
  float* cDec = (float*)carve(HDIM * 4);
  float* hsDec = (float*)carve((size_t)SEQ * HDIM * 4);
  ushort_t* hsBf = (ushort_t*)carve((size_t)SEQ * HDIM * 2);
  float* tgtLogit = (float*)carve(SEQ * 4);
  float* cmax = (float*)carve((size_t)32 * NCHUNK * 16 * 4);
  float* csum = (float*)carve((size_t)32 * NCHUNK * 16 * 4);
  int* bar = (int*)carve(16 * 4);

  // 0) reset barrier state (required for graph replay determinism)
  k_zero_i32<<<1, 32, 0, stream>>>(bar, 16);

  // 1) weight conversions fp32 -> bf16
  k_f32_to_bf16<<<2048, 256, 0, stream>>>(enc_Wih, wihE, G4 * EDIM);
  k_f32_to_bf16<<<2048, 256, 0, stream>>>(enc_Whh, whhE, G4 * HDIM);
  k_f32_to_bf16<<<2048, 256, 0, stream>>>(dec_Wih, wihD, G4 * EDIM);
  k_f32_to_bf16<<<2048, 256, 0, stream>>>(dec_Whh, whhD, G4 * HDIM);
  k_f32_to_bf16<<<8192, 256, 0, stream>>>(W_out, woutB, VTRG * HDIM);

  // 2) embedding gathers (+ bf16, zero-pad)
  k_gather_embed<<<SEQ, 256, 0, stream>>>(enc_emb, source, embE, SEQ);
  k_gather_embed<<<SEQ, 256, 0, stream>>>(dec_emb, target, embD, TDEC); // trg_in = target[:-1]

  // 3) encoder: pre = emb @ Wih.T + (bih+bhh), then sequential scan
  {
    dim3 g(G4 / 128, SEQ / 16);
    k_wmma_gemm_bias<<<g, 256, 0, stream>>>(embE, wihE, enc_bih, enc_bhh, preE, G4);
  }
  k_lstm_scan<<<SCAN_WG, 256, 0, stream>>>(preE, whhE, nullptr, nullptr, SEQ,
                                           nullptr, hEnc, cEnc, hbuf, bar);

  // 4) decoder: pre GEMM, scan seeded with encoder (h, c), record hs
  {
    dim3 g(G4 / 128, SEQ / 16);
    k_wmma_gemm_bias<<<g, 256, 0, stream>>>(embD, wihD, dec_bih, dec_bhh, preD, G4);
  }
  k_lstm_scan<<<SCAN_WG, 256, 0, stream>>>(preD, whhD, hEnc, cEnc, TDEC,
                                           hsDec, hDec, cDec, hbuf, bar + 8);

  // 5) hs -> bf16 (pad row 511)
  k_hs_to_bf16<<<SEQ, 256, 0, stream>>>(hsDec, hsBf, TDEC);

  // 6) fused logits GEMM + online logsumexp (+ target-logit capture)
  {
    dim3 g(NCHUNK, SEQ / 16);
    k_logits_lse<<<g, 256, 0, stream>>>(hsBf, woutB, b_out, target,
                                        cmax, csum, tgtLogit, TDEC);
  }

  // 7) merge chunk stats -> losses
  k_loss_final<<<(TDEC + 255) / 256, 256, 0, stream>>>(cmax, csum, tgtLogit,
                                                       losses, TDEC);
}
